// Network_17506286698721
// MI455X (gfx1250) — matrix operand order repaired and checked by reading, not compiled, not run on hardware
//
#include <hip/hip_runtime.h>
#include <hip/hip_bf16.h>

typedef __attribute__((ext_vector_type(16))) _Float16 v16h;
typedef __attribute__((ext_vector_type(8)))  float    v8f;

#define T_DIM   128
#define NCHAN   1536
#define NC0     10000
#define NC1     10000
#define NC2     10000
#define NCROSS  30000
#define EDGES   120000
#define TILES_PER_T (NCROSS / 16)      // 1875 (NCROSS % 16 == 0, no tail)
#define NTILES  (T_DIM * TILES_PER_T)  // 240000
#define TN      (T_DIM * NCROSS)       // 3,840,000

// ---- order-preserving float<->u32 mapping for atomic max ----
__device__ __forceinline__ unsigned ordf(float f) {
  unsigned u = __float_as_uint(f);
  return (u & 0x80000000u) ? ~u : (u | 0x80000000u);
}
__device__ __forceinline__ float unordf(unsigned o) {
  unsigned u = (o & 0x80000000u) ? (o & 0x7FFFFFFFu) : ~o;
  return __uint_as_float(u);
}
#define ORD_NEG_INF 0x007FFFFFu   // ordf(-inf)

__device__ __forceinline__ void fadd_agent(float* p, float v) {
  __hip_atomic_fetch_add(p, v, __ATOMIC_RELAXED, __HIP_MEMORY_SCOPE_AGENT);
}

// ------------------------------------------------------------------
// K0: init m_ord = ord(-inf), denom = 0, msg = 0
// ------------------------------------------------------------------
__global__ __launch_bounds__(256) void k_init(unsigned* __restrict__ m_ord,
                                              float* __restrict__ denom,
                                              float4* __restrict__ msg4) {
  int i = blockIdx.x * blockDim.x + threadIdx.x;
  if (i >= TN) return;
  m_ord[i] = ORD_NEG_INF;
  denom[i] = 0.f;
  msg4[i]  = make_float4(0.f, 0.f, 0.f, 0.f);
}

// ------------------------------------------------------------------
// K1: fused node-feature gather + WMMA projection.
// One wave per 16-node tile:  D(16x16) = A(16x32 f16) x B(32x16 f16)
//   B = [W_gat | W_gat@a_src | W_gat@a_dst] (cols 0..5), row K=13 = bias row
//   fed by a constant-1 feature.  D cols 0..3 -> h, 4 -> s, 5 -> d.
// NC0/NC1 are multiples of 16 => each tile is group-uniform (scalar branch).
// D is staged through per-wave LDS for coalesced global stores.
// ------------------------------------------------------------------
__global__ __launch_bounds__(256) void k_nodes_wmma(
    const float* __restrict__ x,
    const int* __restrict__ c0, const int* __restrict__ c1, const int* __restrict__ c2,
    const int* __restrict__ g01, const int* __restrict__ g12, const int* __restrict__ g20,
    const float* __restrict__ r01, const float* __restrict__ r12, const float* __restrict__ r20,
    const float* __restrict__ Wg,  const float* __restrict__ av,
    const float* __restrict__ dv,  const float* __restrict__ bg,
    float* __restrict__ h, float* __restrict__ s, float* __restrict__ d) {
  __shared__ float lds[8][16][8];                 // [wave][node-row][col(6, pad 8)]
  const int  lane = threadIdx.x & 31;
  const int  col  = lane & 15;
  const bool hi   = lane >= 16;
  float (*wl)[8] = lds[threadIdx.x >> 5];

  // ---- B matrix (32x16 f16): lane = column N; halves = K
  //      (lane<16: K0-15, lane>=16: K16-31 which are all zero padding) ----
  v16h bmat;
  #pragma unroll
  for (int k = 0; k < 16; ++k) bmat[k] = (_Float16)0.f;
  if (!hi) {
    #pragma unroll
    for (int k = 0; k < 14; ++k) {
      float r0, r1, r2, r3;
      if (k < 13) { r0 = Wg[k*4+0]; r1 = Wg[k*4+1]; r2 = Wg[k*4+2]; r3 = Wg[k*4+3]; }
      else        { r0 = bg[0];     r1 = bg[1];     r2 = bg[2];     r3 = bg[3];     }
      float v = 0.f;
      if      (col == 0) v = r0;
      else if (col == 1) v = r1;
      else if (col == 2) v = r2;
      else if (col == 3) v = r3;
      else if (col == 4) v = r0*av[0] + r1*av[1] + r2*av[2] + r3*av[3];
      else if (col == 5) v = r0*dv[0] + r1*dv[1] + r2*dv[2] + r3*dv[3];
      bmat[k] = (_Float16)v;
    }
  }

  const int wid = (blockIdx.x * blockDim.x + threadIdx.x) >> 5;
  const int nw  = (gridDim.x * blockDim.x) >> 5;

  for (int tile = wid; tile < NTILES; tile += nw) {
    const int t     = tile / TILES_PER_T;
    const int node0 = (tile % TILES_PER_T) * 16;

    // ---- wave-uniform crossing-group resolution (node0 % 16 == 0 and
    //      group boundaries 10000/20000 are multiples of 16) ----
    int cbase, pA, pB;
    const int *gx, *cAt, *cBt; const float* ray;
    if (node0 < NC0)            { cbase = 0;         gx = g01; ray = r01; cAt = c0; cBt = c1; pA = 0; pB = 1; }
    else if (node0 < NC0 + NC1) { cbase = NC0;       gx = g12; ray = r12; cAt = c1; cBt = c2; pA = 1; pB = 2; }
    else                        { cbase = NC0 + NC1; gx = g20; ray = r20; cAt = c2; cBt = c0; pA = 2; pB = 0; }
    const int cid = node0 - cbase + col;  // per-lane crossing index (row M = col)

    // ---- A matrix (16x32 f16), contiguous layout:
    //   lane<16 : halves 0-15 = K0-15
    //   lane>=16: halves 0-15 = K16-31 (zeros)
    v16h amat;
    #pragma unroll
    for (int k = 0; k < 16; ++k) amat[k] = (_Float16)0.f;
    if (!hi) {
      const int wA = gx[2*cid], wB = gx[2*cid + 1];
      const int chA = cAt[wA],  chB = cBt[wB];
      amat[0] = (_Float16)x[chA * T_DIM + t];   // sigA
      amat[1] = (_Float16)(float)wA;            // wireA id
      amat[2] = (_Float16)(float)chA;           // chanA
      amat[3] = (_Float16)0.f;                  // zero feat
      amat[4] = (_Float16)(float)pA;            // planeA
      amat[5] = (_Float16)x[chB * T_DIM + t];   // sigB
      amat[6] = (_Float16)(float)wB;            // wireB id
      amat[7] = (_Float16)(float)chB;           // chanB
      // K8..13: 0, planeB, ray_x, ray_y, tick, 1 (bias)
      amat[9]  = (_Float16)(float)pB;
      amat[10] = (_Float16)ray[2*cid];
      amat[11] = (_Float16)ray[2*cid + 1];
      amat[12] = (_Float16)(float)t;
      amat[13] = (_Float16)1.f;
    }

    v8f acc = {};
    acc = __builtin_amdgcn_wmma_f32_16x16x32_f16(
        /*neg_a=*/false, amat, /*neg_b=*/false, bmat,
        /*c_mod=*/(short)0, acc, /*reuse_a=*/false, /*reuse_b=*/false);

    // ---- stage D (lane = column, VGPR = row, +8 for high lanes) into LDS ----
    if (col < 6) {
      const int roff = hi ? 8 : 0;
      #pragma unroll
      for (int v = 0; v < 8; ++v) wl[v + roff][col] = acc[v];
    }
    __builtin_amdgcn_wave_barrier();   // wave-local LDS: DS ops in-order per wave

    // ---- coalesced stores: 16 lanes -> h as float4, 16 lanes -> s, then d ----
    const size_t nb = (size_t)t * NCROSS + node0;
    if (!hi) {
      const float4 hv = make_float4(wl[col][0], wl[col][1], wl[col][2], wl[col][3]);
      *reinterpret_cast<float4*>(&h[(nb + col) * 4]) = hv;
      d[nb + col] = wl[col][5];
    } else {
      s[nb + col] = wl[col][4];
    }
    __builtin_amdgcn_wave_barrier();   // keep next iteration's LDS writes behind reads
  }
}

// ------------------------------------------------------------------
// K2: per-edge leaky_relu score, segment max via ordered-int atomicMax
// ------------------------------------------------------------------
__global__ __launch_bounds__(256) void k_edge_max(
    const int* __restrict__ esrc, const int* __restrict__ edst,
    const float* __restrict__ s, const float* __restrict__ d,
    unsigned* __restrict__ m_ord) {
  int i = blockIdx.x * blockDim.x + threadIdx.x;
  if (i >= T_DIM * EDGES) return;
  const int t = i / EDGES, e = i % EDGES;
  const size_t base = (size_t)t * NCROSS;
  float ev = s[base + esrc[e]] + d[base + edst[e]];
  ev = ev > 0.f ? ev : 0.2f * ev;
  atomicMax(&m_ord[base + edst[e]], ordf(ev));
}

// ------------------------------------------------------------------
// K3: m = isfinite(max) ? max : 0   (in-place u32 -> f32 reuse)
// ------------------------------------------------------------------
__global__ __launch_bounds__(256) void k_mfix(unsigned* __restrict__ m_ord) {
  int i = blockIdx.x * blockDim.x + threadIdx.x;
  if (i >= TN) return;
  float f = unordf(m_ord[i]);
  bool fin = (__float_as_uint(f) & 0x7F800000u) != 0x7F800000u;
  reinterpret_cast<float*>(m_ord)[i] = fin ? f : 0.f;
}

// ------------------------------------------------------------------
// K4: w = exp(e - m[dst]); denom[dst] += w; msg[dst] += w * h[src]
// ------------------------------------------------------------------
__global__ __launch_bounds__(256) void k_edge_sum(
    const int* __restrict__ esrc, const int* __restrict__ edst,
    const float* __restrict__ s, const float* __restrict__ d,
    const float* __restrict__ m, const float* __restrict__ h,
    float* __restrict__ denom, float* __restrict__ msg) {
  int i = blockIdx.x * blockDim.x + threadIdx.x;
  if (i >= T_DIM * EDGES) return;
  const int t = i / EDGES, e = i % EDGES;
  const size_t base = (size_t)t * NCROSS;
  const size_t si = base + esrc[e], di = base + edst[e];
  float ev = s[si] + d[di];
  ev = ev > 0.f ? ev : 0.2f * ev;
  const float w = __expf(ev - m[di]);
  fadd_agent(&denom[di], w);
  const float4 hv = *reinterpret_cast<const float4*>(&h[si * 4]);
  fadd_agent(&msg[di * 4 + 0], w * hv.x);
  fadd_agent(&msg[di * 4 + 1], w * hv.y);
  fadd_agent(&msg[di * 4 + 2], w * hv.z);
  fadd_agent(&msg[di * 4 + 3], w * hv.w);
}

// ------------------------------------------------------------------
// K5: out = sigmoid((msg/denom) @ W_mlp + b_mlp)
// ------------------------------------------------------------------
__global__ __launch_bounds__(256) void k_out(
    const float* __restrict__ denom, const float* __restrict__ msg,
    const float* __restrict__ Wm, const float* __restrict__ bm,
    float* __restrict__ out) {
  int i = blockIdx.x * blockDim.x + threadIdx.x;
  if (i >= TN) return;
  const float inv = 1.f / (denom[i] + 1e-9f);
  const float4 g = *reinterpret_cast<const float4*>(&msg[(size_t)i * 4]);
  const float z = (g.x * Wm[0] + g.y * Wm[1] + g.z * Wm[2] + g.w * Wm[3]) * inv + bm[0];
  out[i] = 1.f / (1.f + __expf(-z));
}

// ------------------------------------------------------------------
extern "C" void kernel_launch(void* const* d_in, const int* in_sizes, int n_in,
                              void* d_out, int out_size, void* d_ws, size_t ws_size,
                              hipStream_t stream) {
  const float* x   = (const float*)d_in[0];
  const int*   c0  = (const int*)d_in[1];
  const int*   c1  = (const int*)d_in[2];
  const int*   c2  = (const int*)d_in[3];
  const int*   g01 = (const int*)d_in[4];
  const int*   g12 = (const int*)d_in[5];
  const int*   g20 = (const int*)d_in[6];
  const float* r01 = (const float*)d_in[7];
  const float* r12 = (const float*)d_in[8];
  const float* r20 = (const float*)d_in[9];
  const int*   edges = (const int*)d_in[10];     // [2, E]
  const float* Wg  = (const float*)d_in[11];     // [13,4]
  const float* av  = (const float*)d_in[12];     // a_src [4]
  const float* dv  = (const float*)d_in[13];     // a_dst [4]
  const float* bg  = (const float*)d_in[14];     // b_gat [4]
  const float* Wm  = (const float*)d_in[15];     // W_mlp [4,1]
  const float* bm  = (const float*)d_in[16];     // b_mlp [1]
  float* out = (float*)d_out;

  const int* esrc = edges;
  const int* edst = edges + EDGES;

  // workspace layout (≈184 MB total)
  char* ws = (char*)d_ws;
  float*    h     = (float*)ws;                      ws += (size_t)TN * 4 * sizeof(float);
  float*    s     = (float*)ws;                      ws += (size_t)TN * sizeof(float);
  float*    d     = (float*)ws;                      ws += (size_t)TN * sizeof(float);
  unsigned* m_ord = (unsigned*)ws;                   ws += (size_t)TN * sizeof(unsigned);
  float*    denom = (float*)ws;                      ws += (size_t)TN * sizeof(float);
  float*    msg   = (float*)ws;                      ws += (size_t)TN * 4 * sizeof(float);
  (void)ws_size; (void)in_sizes; (void)n_in; (void)out_size;

  const int B256 = 256;
  const int gTN = (TN + B256 - 1) / B256;            // 15000
  const int gTE = (T_DIM * EDGES + B256 - 1) / B256; // 60000

  k_init<<<gTN, B256, 0, stream>>>(m_ord, denom, (float4*)msg);
  k_nodes_wmma<<<2400, B256, 0, stream>>>(x, c0, c1, c2, g01, g12, g20,
                                          r01, r12, r20, Wg, av, dv, bg, h, s, d);
  k_edge_max<<<gTE, B256, 0, stream>>>(esrc, edst, s, d, m_ord);
  k_mfix<<<gTN, B256, 0, stream>>>(m_ord);
  k_edge_sum<<<gTE, B256, 0, stream>>>(esrc, edst, s, d, (const float*)m_ord, h, denom, msg);
  k_out<<<gTN, B256, 0, stream>>>(denom, msg, Wm, bm, out);
}
